// Attention_56126632624586
// MI455X (gfx1250) — matrix operand order repaired and checked by reading, not compiled, not run on hardware
//
#include <hip/hip_runtime.h>

typedef __attribute__((ext_vector_type(16))) _Float16 v16h;
typedef __attribute__((ext_vector_type(8)))  _Float16 v8h;
typedef __attribute__((ext_vector_type(8)))  float    v8f;

#define WMMA_F16(a,b,c) __builtin_amdgcn_wmma_f32_16x16x32_f16( \
    false,(a),false,(b),(short)0,(c),false,false)

#define NTOK  2048
#define DHEAD 32
#define HEADS 8
#define CDIM  256
#define QK_SCALE 0.17677669529663687f  /* 32^-0.5 */

// ---- fragment loaders (contiguous k = 16g+i, same order in A and B) ------------

// A-matrix 16x32 f16, row-major memory [M][K] with leading dim ld.
// lane l: M = l&15, g = l>>4; element i -> K = 16g+i.
__device__ __forceinline__ v16h load_a_rm(const _Float16* p, int ld, int lane) {
  int m = lane & 15, g = lane >> 4;
  const _Float16* r = p + (size_t)m * ld + 16 * g;
  v8h lo = *(const v8h*)r;
  v8h hi = *(const v8h*)(r + 8);
  return __builtin_shufflevector(lo, hi, 0,1,2,3,4,5,6,7,8,9,10,11,12,13,14,15);
}

// B-matrix 32x16 f16 from memory laid out [N][K] (i.e. B-transposed rows),
// leading dim ld. lane l: N = l&15, element i -> K = 16*(l>>4) + i.
// One contiguous 32-byte load per lane.
__device__ __forceinline__ v16h load_b_nk(const _Float16* p, int ld, int lane) {
  int n = lane & 15, g = lane >> 4;
  return *(const v16h*)(p + (size_t)n * ld + 16 * g);
}

// ---- fp32 -> f16 conversion ------------------------------------------------
__global__ void cvt_kernel(const float* __restrict__ src,
                           _Float16* __restrict__ dst, int nElem) {
  int i = blockIdx.x * 256 + threadIdx.x;
  if (i < nElem) dst[i] = (_Float16)src[i];
}

// ---- QKV projection: [8192,256] x [768,256]^T, scatter to Q/K/Vt -----------
__global__ __launch_bounds__(256) void qkv_kernel(
    const _Float16* __restrict__ Xh, const _Float16* __restrict__ Wh,
    _Float16* __restrict__ Qf, _Float16* __restrict__ Kf,
    _Float16* __restrict__ Vt) {
  int lane = threadIdx.x & 31;
  int w    = threadIdx.x >> 5;
  int row0 = blockIdx.x * 16;             // 512 row tiles of 8192 rows
  int col0 = blockIdx.y * 128 + w * 16;   // 768 cols, y in 0..5
  v8f acc = {};
#pragma unroll
  for (int kb = 0; kb < CDIM; kb += 32) {
    v16h af = load_a_rm(Xh + (size_t)row0 * CDIM + kb, CDIM, lane);
    v16h bf = load_b_nk(Wh + (size_t)col0 * CDIM + kb, CDIM, lane);
    acc = WMMA_F16(af, bf, acc);
  }
  int n = lane & 15, g = lane >> 4;
  int j = col0 + n;
  int which = j >> 8;          // 0=Q 1=K 2=V
  int c = j & 255;
  int h = c >> 5, d = c & 31;
#pragma unroll
  for (int r = 0; r < 8; r++) {
    int row = row0 + r + 8 * g;
    int b = row >> 11, tok = row & 2047;
    float v = acc[r];
    if (which == 0) {
      Qf[((size_t)(b * HEADS + h) * NTOK + tok) * DHEAD + d] =
          (_Float16)(v * QK_SCALE);
    } else if (which == 1) {
      Kf[((size_t)(b * HEADS + h) * NTOK + tok) * DHEAD + d] = (_Float16)v;
    } else {
      // V transposed: [BH, D, N] so PV B-fragments are contiguous loads
      Vt[((size_t)(b * HEADS + h) * DHEAD + d) * NTOK + tok] = (_Float16)v;
    }
  }
}

// ---- flash attention: 8 waves/block, 16 query rows per wave ----------------
__global__ __launch_bounds__(256) void attn_kernel(
    const _Float16* __restrict__ Qf, const _Float16* __restrict__ Kf,
    const _Float16* __restrict__ Vt, _Float16* __restrict__ AttnO) {
  __shared__ _Float16 Pbuf[8][16 * 32];   // per-wave P staging (C->A relayout)
  int lane = threadIdx.x & 31;
  int w    = threadIdx.x >> 5;
  int bh   = blockIdx.x >> 4;             // b*8+h, 0..31
  int mt   = blockIdx.x & 15;             // 128-query tile
  int b = bh >> 3, h = bh & 7;
  int qrow0 = mt * 128 + w * 16;
  int n = lane & 15, g = lane >> 4;

  // Q A-fragment stays in registers for the whole key loop (scale pre-folded)
  v16h qa = load_a_rm(Qf + ((size_t)bh * NTOK + qrow0) * DHEAD, DHEAD, lane);

  float mrow[8], lrow[8];
#pragma unroll
  for (int r = 0; r < 8; r++) { mrow[r] = -1e30f; lrow[r] = 0.f; }
  v8f o0 = {};  // output cols d = 0..15
  v8f o1 = {};  // output cols d = 16..31

  for (int j = 0; j < NTOK; j += 32) {
    // --- scores for 32 keys: two 16x16x32 WMMAs (K-dim == D == 32) ---
    const _Float16* Kb = Kf + ((size_t)bh * NTOK + j) * DHEAD;
    v16h kb0 = load_b_nk(Kb, DHEAD, lane);
    v16h kb1 = load_b_nk(Kb + 16 * DHEAD, DHEAD, lane);
    v8f z = {};
    v8f s0 = WMMA_F16(qa, kb0, z);
    v8f s1 = WMMA_F16(qa, kb1, z);

    // --- online softmax per row (row = r + 8g, spread over 16-lane group) ---
#pragma unroll
    for (int r = 0; r < 8; r++) {
      float mx = fmaxf(s0[r], s1[r]);
      mx = fmaxf(mx, __shfl_xor(mx, 1));
      mx = fmaxf(mx, __shfl_xor(mx, 2));
      mx = fmaxf(mx, __shfl_xor(mx, 4));
      mx = fmaxf(mx, __shfl_xor(mx, 8));
      float mnew = fmaxf(mrow[r], mx);
      float p0 = __expf(s0[r] - mnew);
      float p1 = __expf(s1[r] - mnew);
      float corr = __expf(mrow[r] - mnew);
      mrow[r] = mnew;
      float ps = p0 + p1;
      ps += __shfl_xor(ps, 1);
      ps += __shfl_xor(ps, 2);
      ps += __shfl_xor(ps, 4);
      ps += __shfl_xor(ps, 8);
      lrow[r] = lrow[r] * corr + ps;
      o0[r] *= corr;
      o1[r] *= corr;
      // stage P row-major [m][k] for A-frag reload
      Pbuf[w][(r + 8 * g) * 32 + n]      = (_Float16)p0;
      Pbuf[w][(r + 8 * g) * 32 + 16 + n] = (_Float16)p1;
    }
    __syncthreads();

    // --- P(16x32) x V(32x16) for both halves of D ---
    v16h pa = load_a_rm(&Pbuf[w][0], 32, lane);
    const _Float16* Vb = Vt + (size_t)bh * DHEAD * NTOK + j;
    v16h vb0 = load_b_nk(Vb, NTOK, lane);              // d = 0..15
    v16h vb1 = load_b_nk(Vb + 16 * NTOK, NTOK, lane);  // d = 16..31
    o0 = WMMA_F16(pa, vb0, o0);
    o1 = WMMA_F16(pa, vb1, o1);
    __syncthreads();
  }

  // normalize and store as [B, N, C] (heads concatenated) in f16
#pragma unroll
  for (int r = 0; r < 8; r++) {
    float inv = 1.0f / lrow[r];
    int tok = qrow0 + r + 8 * g;
    _Float16* dst = AttnO + ((size_t)(b * NTOK + tok)) * CDIM + h * DHEAD;
    dst[n]      = (_Float16)(o0[r] * inv);
    dst[16 + n] = (_Float16)(o1[r] * inv);
  }
}

// ---- output projection: [8192,256] x [256,256]^T + bias, fp32 out ----------
__global__ __launch_bounds__(256) void proj_kernel(
    const _Float16* __restrict__ A, const _Float16* __restrict__ Wh,
    const float* __restrict__ bias, float* __restrict__ out) {
  int lane = threadIdx.x & 31;
  int w    = threadIdx.x >> 5;
  int row0 = blockIdx.x * 16;
  int col0 = blockIdx.y * 128 + w * 16;   // y in 0..1
  v8f acc = {};
#pragma unroll
  for (int kb = 0; kb < CDIM; kb += 32) {
    v16h af = load_a_rm(A + (size_t)row0 * CDIM + kb, CDIM, lane);
    v16h bf = load_b_nk(Wh + (size_t)col0 * CDIM + kb, CDIM, lane);
    acc = WMMA_F16(af, bf, acc);
  }
  int n = lane & 15, g = lane >> 4;
  float bv = bias[col0 + n];
#pragma unroll
  for (int r = 0; r < 8; r++) {
    out[(size_t)(row0 + r + 8 * g) * CDIM + col0 + n] = acc[r] + bv;
  }
}

extern "C" void kernel_launch(void* const* d_in, const int* in_sizes, int n_in,
                              void* d_out, int out_size, void* d_ws,
                              size_t ws_size, hipStream_t stream) {
  (void)in_sizes; (void)n_in; (void)out_size; (void)ws_size;
  const float* x      = (const float*)d_in[0];  // [4,2048,256]
  const float* qkv_w  = (const float*)d_in[1];  // [768,256]
  const float* proj_w = (const float*)d_in[2];  // [256,256]
  const float* proj_b = (const float*)d_in[3];  // [256]
  float* out = (float*)d_out;                   // [4,2048,256]

  const int NX = 4 * NTOK * CDIM;      // 2097152
  const int NQW = 3 * CDIM * CDIM;     // 196608
  const int NPW = CDIM * CDIM;         // 65536

  char* ws = (char*)d_ws;
  size_t off = 0;
  auto alloc = [&](size_t bytes) {
    void* p = ws + off;
    off += (bytes + 255) & ~(size_t)255;
    return p;
  };
  _Float16* Xh    = (_Float16*)alloc((size_t)NX * 2);
  _Float16* QWh   = (_Float16*)alloc((size_t)NQW * 2);
  _Float16* PWh   = (_Float16*)alloc((size_t)NPW * 2);
  _Float16* Qf    = (_Float16*)alloc((size_t)NX * 2);
  _Float16* Kf    = (_Float16*)alloc((size_t)NX * 2);
  _Float16* Vt    = (_Float16*)alloc((size_t)NX * 2);
  _Float16* AttnO = (_Float16*)alloc((size_t)NX * 2);

  cvt_kernel<<<NX / 256, 256, 0, stream>>>(x, Xh, NX);
  cvt_kernel<<<NQW / 256, 256, 0, stream>>>(qkv_w, QWh, NQW);
  cvt_kernel<<<NPW / 256, 256, 0, stream>>>(proj_w, PWh, NPW);

  // 8192 rows / 16 = 512 row tiles; 768 cols = 6 groups of 128 (8 waves x 16)
  qkv_kernel<<<dim3(512, 6), 256, 0, stream>>>(Xh, QWh, Qf, Kf, Vt);
  // 32 (b,h) pairs x 16 query tiles of 128 rows
  attn_kernel<<<512, 256, 0, stream>>>(Qf, Kf, Vt, AttnO);
  // 512 row tiles x 2 col groups of 128
  proj_kernel<<<dim3(512, 2), 256, 0, stream>>>(AttnO, PWh, proj_b, out);
}
